// GNNModel0_34239479283770
// MI455X (gfx1250) — matrix operand order repaired and checked by reading, not compiled, not run on hardware
//
#include <hip/hip_runtime.h>
#include <hip/hip_bf16.h>

typedef __attribute__((ext_vector_type(16))) __bf16 v16bf;
typedef __attribute__((ext_vector_type(8)))  __bf16 v8bf;
typedef __attribute__((ext_vector_type(8)))  float  v8f;

#define NODES  100000
#define NGRAPH 64
#define NCLS   38

__global__ __launch_bounds__(256) void k_zero(float* p, int n) {
  int i = blockIdx.x * 256 + threadIdx.x;
  if (i < n) p[i] = 0.f;
}

__global__ __launch_bounds__(256) void k_deg(const int* __restrict__ dst, float* __restrict__ deg, int E) {
  int i = blockIdx.x * 256 + threadIdx.x;
  if (i < E) atomicAdd(&deg[dst[i]], 1.0f);
}

__global__ __launch_bounds__(256) void k_dinv(const float* __restrict__ deg, float* __restrict__ dinv, int n) {
  int i = blockIdx.x * 256 + threadIdx.x;
  if (i < n) dinv[i] = rsqrtf(deg[i] + 1.0f);   // +1 for self loop
}

// OUT[M,N] = A[M,K] @ W[K,N]  (f32 in, bf16 WMMA, f32 accumulate)
// Fused epilogue: ACC[r,c] = OUT[r,c] * dinv[r]^2  (self-loop message init).
// Requires M%32==0, N%64==0, K%32==0, K<=256. ACC must not alias A.
// Block = 256 threads (8 waves). Block tile: 32 rows x 64 cols; one 16x16 tile per wave.
__global__ __launch_bounds__(256) void k_gemm_wmma(
    const float* __restrict__ A, const float* __restrict__ W,
    float* __restrict__ OUT, float* __restrict__ ACC,
    const float* __restrict__ dinv, int M, int K, int N) {
  __shared__ __align__(16) __bf16 ldsB[64 * 264];   // 64 cols x (K+8) pad, max K=256
  const int Kpad = K + 8;
  const int tid = threadIdx.x;
  const int colBase = blockIdx.x * 64;
  const int rowBase = blockIdx.y * 32;

  // Stage B tile column-major (bf16) so each lane's fragment is contiguous 16B chunks.
  for (int idx = tid; idx < (K << 6); idx += 256) {
    int n = idx & 63, k = idx >> 6;
    ldsB[n * Kpad + k] = (__bf16)W[k * N + colBase + n];
  }
  __syncthreads();

  const int wave = tid >> 5, lane = tid & 31;
  const int mt = wave >> 2, nt = wave & 3;
  const int hlf = lane >> 4, l16 = lane & 15;

  const float* arow = A + (long)(rowBase + mt * 16 + l16) * K;
  const int kb = hlf * 8;                       // A half-wave K sub-offset
  const __bf16* bcol = ldsB + (nt * 16 + l16) * Kpad;

  v8f acc = {};
  for (int k0 = 0; k0 < K; k0 += 32) {
    // A fragment: lanes 0-15 hold K {k0..k0+7, k0+16..k0+23}; lanes 16-31 shifted by 8
    const float* ap = arow + k0 + kb;
    float4 a0 = *(const float4*)(ap);
    float4 a1 = *(const float4*)(ap + 4);
    float4 a2 = *(const float4*)(ap + 16);
    float4 a3 = *(const float4*)(ap + 20);
    v16bf av;
    av[0]=(__bf16)a0.x;  av[1]=(__bf16)a0.y;  av[2]=(__bf16)a0.z;  av[3]=(__bf16)a0.w;
    av[4]=(__bf16)a1.x;  av[5]=(__bf16)a1.y;  av[6]=(__bf16)a1.z;  av[7]=(__bf16)a1.w;
    av[8]=(__bf16)a2.x;  av[9]=(__bf16)a2.y;  av[10]=(__bf16)a2.z; av[11]=(__bf16)a2.w;
    av[12]=(__bf16)a3.x; av[13]=(__bf16)a3.y; av[14]=(__bf16)a3.z; av[15]=(__bf16)a3.w;
    // B fragment: lane = column, half-wave 0 holds K {k0..k0+7, k0+16..k0+23}; half-wave 1 shifted by 8
    v8bf b0 = *(const v8bf*)(bcol + k0 + hlf * 8);
    v8bf b1 = *(const v8bf*)(bcol + k0 + hlf * 8 + 16);
    v16bf bv = __builtin_shufflevector(b0, b1, 0,1,2,3,4,5,6,7,8,9,10,11,12,13,14,15);
    acc = __builtin_amdgcn_wmma_f32_16x16x32_bf16(false, av, false, bv,
                                                  (short)0, acc, false, false);
  }

  const int col = colBase + nt * 16 + l16;
  const int r0  = rowBase + mt * 16 + hlf * 8;   // C/D: VGPR j -> row r0+j
  #pragma unroll
  for (int j = 0; j < 8; ++j) {
    float v = acc[j];
    float d = dinv[r0 + j];
    long off = (long)(r0 + j) * N + col;
    OUT[off] = v;
    ACC[off] = v * d * d;   // self-loop message (also initializes the scatter buffer)
  }
}

// acc[dst] += h[src] * dinv[src]*dinv[dst]   (edge-parallel, 4 feats per thread)
__global__ __launch_bounds__(256) void k_scatter(
    const float* __restrict__ h, const int* __restrict__ src,
    const int* __restrict__ dst, const float* __restrict__ dinv,
    float* __restrict__ acc, int E, int cshift, int F) {
  long t = (long)blockIdx.x * 256 + threadIdx.x;
  int e = (int)(t >> cshift);
  if (e >= E) return;
  int c = ((int)t & ((1 << cshift) - 1)) << 2;
  int s = src[e], d = dst[e];
  float norm = dinv[s] * dinv[d];
  float4 hv = *(const float4*)(h + (long)s * F + c);
  float* ap = acc + (long)d * F + c;
  atomicAdd(ap + 0, hv.x * norm);
  atomicAdd(ap + 1, hv.y * norm);
  atomicAdd(ap + 2, hv.z * norm);
  atomicAdd(ap + 3, hv.w * norm);
}

__global__ __launch_bounds__(256) void k_bias_relu(
    float* __restrict__ acc, const float* __restrict__ bias, int fmask, long n4) {
  long i = (long)blockIdx.x * 256 + threadIdx.x;
  if (i >= n4) return;
  long idx = i << 2;
  int f = (int)idx & fmask;
  float4 b = *(const float4*)(bias + f);
  float4 v = *(const float4*)(acc + idx);
  v.x = fmaxf(v.x + b.x, 0.f);
  v.y = fmaxf(v.y + b.y, 0.f);
  v.z = fmaxf(v.z + b.z, 0.f);
  v.w = fmaxf(v.w + b.w, 0.f);
  *(float4*)(acc + idx) = v;
}

// Layer-2 finalize fused with global_add_pool:
// v = relu(acc + b2); acc = v; pooled[batch[node]] += v   (F = 128)
__global__ __launch_bounds__(256) void k_bias_relu_pool(
    float* __restrict__ acc, const float* __restrict__ bias,
    const int* __restrict__ batch, float* __restrict__ pooled, long n4) {
  long i = (long)blockIdx.x * 256 + threadIdx.x;
  if (i >= n4) return;
  long idx = i << 2;
  int node = (int)(idx >> 7);
  int f = (int)idx & 127;
  float4 b = *(const float4*)(bias + f);
  float4 v = *(const float4*)(acc + idx);
  v.x = fmaxf(v.x + b.x, 0.f);
  v.y = fmaxf(v.y + b.y, 0.f);
  v.z = fmaxf(v.z + b.z, 0.f);
  v.w = fmaxf(v.w + b.w, 0.f);
  *(float4*)(acc + idx) = v;
  int g = batch[node];
  float* pp = pooled + g * 128 + f;
  atomicAdd(pp + 0, v.x);
  atomicAdd(pp + 1, v.y);
  atomicAdd(pp + 2, v.z);
  atomicAdd(pp + 3, v.w);
}

// pooled2 = pooled @ Wx + bx   (64x128 @ 128x128)
__global__ __launch_bounds__(256) void k_head1(
    const float* __restrict__ pooled, const float* __restrict__ Wx,
    const float* __restrict__ bx, float* __restrict__ pooled2) {
  int t = blockIdx.x * 256 + threadIdx.x;
  if (t >= NGRAPH * 128) return;
  int g = t >> 7, f = t & 127;
  float s = bx[f];
  const float* pr = pooled + g * 128;
  for (int k = 0; k < 128; ++k) s += pr[k] * Wx[k * 128 + f];
  pooled2[t] = s;
}

// logits = pooled2 @ Wfc + bfc; out = log_softmax(logits)
__global__ __launch_bounds__(64) void k_head2(
    const float* __restrict__ pooled2, const float* __restrict__ Wfc,
    const float* __restrict__ bfc, float* __restrict__ out) {
  __shared__ float sl[NCLS];
  __shared__ float slse;
  int g = blockIdx.x, c = threadIdx.x;
  if (c < NCLS) {
    float s = bfc[c];
    const float* pr = pooled2 + g * 128;
    for (int k = 0; k < 128; ++k) s += pr[k] * Wfc[k * NCLS + c];
    sl[c] = s;
  }
  __syncthreads();
  if (c == 0) {
    float m = sl[0];
    for (int j = 1; j < NCLS; ++j) m = fmaxf(m, sl[j]);
    float ssum = 0.f;
    for (int j = 0; j < NCLS; ++j) ssum += __expf(sl[j] - m);
    slse = m + __logf(ssum);
  }
  __syncthreads();
  if (c < NCLS) out[g * NCLS + c] = sl[c] - slse;
}

extern "C" void kernel_launch(void* const* d_in, const int* in_sizes, int n_in,
                              void* d_out, int out_size, void* d_ws, size_t ws_size,
                              hipStream_t stream) {
  const float* x    = (const float*)d_in[0];
  const int*   ei   = (const int*)d_in[1];
  const int*   bat  = (const int*)d_in[2];
  // d_in[3] img_feature, d_in[8] Wx0, d_in[9] bx0: computed-but-unused in reference
  const float* W1  = (const float*)d_in[4];
  const float* b1  = (const float*)d_in[5];
  const float* W2  = (const float*)d_in[6];
  const float* b2  = (const float*)d_in[7];
  const float* Wx  = (const float*)d_in[10];
  const float* bx  = (const float*)d_in[11];
  const float* Wfc = (const float*)d_in[12];
  const float* bfc = (const float*)d_in[13];
  float* out = (float*)d_out;

  const int E = in_sizes[1] / 2;
  const int* src = ei;
  const int* dst = ei + E;

  // Workspace layout (bytes)
  const size_t SA = (size_t)NODES * 256 * 4;   // 102.4 MB
  const size_t SC = (size_t)NODES * 128 * 4;   // 51.2 MB
  char* ws = (char*)d_ws;
  float* buffA   = (float*)(ws);               // h (GEMM out): layer1 256-wide, layer2 128-wide
  float* buffB   = (float*)(ws + SA);          // layer-1 acc / layer-2 GEMM input
  float* buffC   = (float*)(ws + 2 * SA);      // layer-2 acc (must not alias layer-2 GEMM input)
  float* deg     = (float*)(ws + 2 * SA + SC);
  float* dinv    = (float*)(ws + 2 * SA + SC + (size_t)NODES * 4);
  float* pooled  = (float*)(ws + 2 * SA + SC + (size_t)NODES * 8);
  float* pooled2 = pooled + NGRAPH * 128;

  // Degree + normalization
  k_zero<<<(NODES + 255) / 256, 256, 0, stream>>>(deg, NODES);
  k_zero<<<(NGRAPH * 128 + 255) / 256, 256, 0, stream>>>(pooled, NGRAPH * 128);
  k_deg<<<(E + 255) / 256, 256, 0, stream>>>(dst, deg, E);
  k_dinv<<<(NODES + 255) / 256, 256, 0, stream>>>(deg, dinv, NODES);

  // ---- Layer 1: h1 = x @ W1 (+fused acc init) ; scatter ; +b1 ; relu  (F = 256) ----
  k_gemm_wmma<<<dim3(256 / 64, NODES / 32), 256, 0, stream>>>(
      x, W1, buffA, buffB, dinv, NODES, 128, 256);
  long n4_1 = (long)NODES * 256 / 4;
  long ts1 = (long)E * 64;   // E * (256/4) threads
  k_scatter<<<(unsigned)((ts1 + 255) / 256), 256, 0, stream>>>(buffA, src, dst, dinv, buffB, E, 6, 256);
  k_bias_relu<<<(unsigned)((n4_1 + 255) / 256), 256, 0, stream>>>(buffB, b1, 255, n4_1);

  // ---- Layer 2: h2 = h1 @ W2 (+fused acc init) ; scatter ; +b2 ; relu ; pool  (F = 128) ----
  k_gemm_wmma<<<dim3(128 / 64, NODES / 32), 256, 0, stream>>>(
      buffB, W2, buffA, buffC, dinv, NODES, 256, 128);
  long n4_2 = (long)NODES * 128 / 4;
  long ts2 = (long)E * 32;   // E * (128/4) threads
  k_scatter<<<(unsigned)((ts2 + 255) / 256), 256, 0, stream>>>(buffA, src, dst, dinv, buffC, E, 5, 128);
  k_bias_relu_pool<<<(unsigned)((n4_2 + 255) / 256), 256, 0, stream>>>(buffC, b2, bat, pooled, n4_2);

  // ---- Head ----
  k_head1<<<(NGRAPH * 128 + 255) / 256, 256, 0, stream>>>(pooled, Wx, bx, pooled2);
  k_head2<<<NGRAPH, 64, 0, stream>>>(pooled2, Wfc, bfc, out);
}
